// DPQNetwork_70239895158853
// MI455X (gfx1250) — compile-verified, matrix operand order checked by reading, not run on hardware
//
#include <hip/hip_runtime.h>
#include <hip/hip_bf16.h>

// Problem constants (match reference)
#define B_TOT 32768
#define C_N   32
#define K_N   256
#define DS    16
#define DIN   512   // C_N * DS
#define DOUT  512

#define ROWS_PER_BLOCK 64   // 4 M-tiles of 16
#define MT_N 4

typedef __attribute__((ext_vector_type(16))) __bf16 v16bf;
typedef __attribute__((ext_vector_type(8)))  float  v8f;

union V16 {
    v16bf v;
    uint4 q[2];
    __bf16 h[16];
};

// ---------------------------------------------------------------------------
// Prep: W (f32, row-major DINxDOUT) -> Wt (bf16, transposed DOUTxDIN)
//       centroids (f32) -> cb (bf16, same layout)
// ---------------------------------------------------------------------------
__global__ void dpq_prep_kernel(const float* __restrict__ W,
                                const float* __restrict__ cent,
                                __bf16* __restrict__ Wt,
                                __bf16* __restrict__ cb) {
    int idx = blockIdx.x * blockDim.x + threadIdx.x;
    if (idx < DIN * DOUT) {
        int kk = idx >> 9;       // row of W (contraction index)
        int n  = idx & 511;      // col of W (output index)
        Wt[n * DIN + kk] = (__bf16)W[idx];
    } else {
        int j = idx - DIN * DOUT;
        if (j < C_N * K_N * DS) cb[j] = (__bf16)cent[j];
    }
}

// ---------------------------------------------------------------------------
// Fused: response GEMM (bf16 WMMA, K padded 16->32) + max/argmax + gather to
// LDS + (64x512)@(512x512) GEMM (bf16 WMMA) -> product.
// One block = 64 b-rows (4 M-tiles), 8 wave32s, B-operand reused across M-tiles.
// ---------------------------------------------------------------------------
#define ACT_STRIDE 520  // 512 + 8 bf16: rotates LDS banks by 4 per row

__global__ __launch_bounds__(256)
void dpq_main_kernel(const float*  __restrict__ inputs,   // (B, C, DS) f32
                     const __bf16* __restrict__ cb,       // (C, K, DS) bf16
                     const __bf16* __restrict__ Wt,       // (DOUT, DIN) bf16
                     float* __restrict__ product,         // (B, DOUT)
                     float* __restrict__ nmse,            // (B, C)  = -max
                     float* __restrict__ codes) {         // (B, C)  argmax as float
    extern __shared__ __align__(16) __bf16 act[];         // [64][ACT_STRIDE]

    const int tid   = threadIdx.x;
    const int lane  = tid & 31;
    const int wave  = tid >> 5;
    const int half  = lane >> 4;   // 0: lanes 0-15, 1: lanes 16-31
    const int l16   = lane & 15;
    const int dbase = half * 8;    // bf16 A/B layout: low lanes K 0..7, high lanes K 8..15
    const int b0    = blockIdx.x * ROWS_PER_BLOCK;

    const uint4 zero4 = make_uint4(0u, 0u, 0u, 0u);

    // ---------------- Phase 1: response + argmax + gather ----------------
    for (int ci = 0; ci < 4; ++ci) {
        const int c = wave * 4 + ci;

        // A tiles for all 4 M-tiles: rows M = l16 of inputs[b0+mt*16+M, c, :].
        V16 a[MT_N];
#pragma unroll
        for (int mt = 0; mt < MT_N; ++mt) {
            const float* ap =
                inputs + ((size_t)(b0 + mt * 16 + l16) * C_N + c) * DS + dbase;
            float4 f0 = *(const float4*)(ap);
            float4 f1 = *(const float4*)(ap + 4);
            a[mt].h[0] = (__bf16)f0.x; a[mt].h[1] = (__bf16)f0.y;
            a[mt].h[2] = (__bf16)f0.z; a[mt].h[3] = (__bf16)f0.w;
            a[mt].h[4] = (__bf16)f1.x; a[mt].h[5] = (__bf16)f1.y;
            a[mt].h[6] = (__bf16)f1.z; a[mt].h[7] = (__bf16)f1.w;
            a[mt].q[1] = zero4;  // K = 16..31 zero padding
        }

        float bestv[MT_N][8];
        int   bestk[MT_N][8];
#pragma unroll
        for (int mt = 0; mt < MT_N; ++mt)
#pragma unroll
            for (int r = 0; r < 8; ++r) {
                bestv[mt][r] = -__builtin_inff();
                bestk[mt][r] = 0;
            }

        for (int kt = 0; kt < 16; ++kt) {
            const int k0 = kt * 16;
            // B tile (shared by all 4 M-tiles): column N = l16 -> centroid row.
            const __bf16* bp = cb + ((size_t)(c * K_N + k0 + l16) * DS + dbase);
            V16 bm;
            bm.q[0] = *(const uint4*)bp;
            bm.q[1] = zero4;  // A's zeros make these K rows dead; keep finite

            const int k = k0 + l16;
#pragma unroll
            for (int mt = 0; mt < MT_N; ++mt) {
                v8f acc = {0.f, 0.f, 0.f, 0.f, 0.f, 0.f, 0.f, 0.f};
                acc = __builtin_amdgcn_wmma_f32_16x16x32_bf16(
                    false, a[mt].v, false, bm.v, (short)0, acc, false, false);
#pragma unroll
                for (int r = 0; r < 8; ++r) {
                    float v  = acc[r];
                    bool  gt = v > bestv[mt][r];
                    bestv[mt][r] = gt ? v : bestv[mt][r];
                    bestk[mt][r] = gt ? k : bestk[mt][r];
                }
            }
        }

        // Cross-lane argmax within each 16-lane half (rows r / r+8).
#pragma unroll
        for (int m = 8; m >= 1; m >>= 1) {
#pragma unroll
            for (int mt = 0; mt < MT_N; ++mt)
#pragma unroll
                for (int r = 0; r < 8; ++r) {
                    float ov = __shfl_xor(bestv[mt][r], m, 16);
                    int   ok = __shfl_xor(bestk[mt][r], m, 16);
                    if (ov > bestv[mt][r] ||
                        (ov == bestv[mt][r] && ok < bestk[mt][r])) {
                        bestv[mt][r] = ov;
                        bestk[mt][r] = ok;
                    }
                }
        }

        // Outputs + gather winning centroid rows into LDS activation tile.
#pragma unroll
        for (int mt = 0; mt < MT_N; ++mt)
#pragma unroll
            for (int r = 0; r < 8; ++r) {
                const int row = mt * 16 + r + 8 * half;  // 0..63 within b-tile
                const int k   = bestk[mt][r];            // uniform across the half
                act[row * ACT_STRIDE + c * DS + l16] =
                    cb[((size_t)c * K_N + k) * DS + l16];
                if (l16 == r) {
                    const size_t o = (size_t)(b0 + row) * C_N + c;
                    nmse[o]  = -bestv[mt][r];
                    codes[o] = (float)k;
                }
            }
    }

    __syncthreads();

    // ---------------- Phase 2: (64 x 512) @ Wt^T -> (64 x 512) ----------------
    for (int j = 0; j < 4; ++j) {
        const int n0 = (wave * 4 + j) * 16;

        v8f acc[MT_N];
#pragma unroll
        for (int mt = 0; mt < MT_N; ++mt)
            acc[mt] = (v8f){0.f, 0.f, 0.f, 0.f, 0.f, 0.f, 0.f, 0.f};

        for (int kc = 0; kc < 16; ++kc) {
            const int kk0 = kc * 32;

            // B from Wt (L2-resident), reused by 4 M-tiles.
            const __bf16* brow = Wt + (size_t)(n0 + l16) * DIN + kk0 + dbase;
            V16 bm;
            bm.q[0] = *(const uint4*)(brow);
            bm.q[1] = *(const uint4*)(brow + 16);

#pragma unroll
            for (int mt = 0; mt < MT_N; ++mt) {
                const __bf16* arow =
                    act + (mt * 16 + l16) * ACT_STRIDE + kk0 + dbase;
                V16 av;
                av.q[0] = *(const uint4*)(arow);
                av.q[1] = *(const uint4*)(arow + 16);
                acc[mt] = __builtin_amdgcn_wmma_f32_16x16x32_bf16(
                    false, av.v, false, bm.v, (short)0, acc[mt], false, false);
            }
        }

#pragma unroll
        for (int mt = 0; mt < MT_N; ++mt)
#pragma unroll
            for (int r = 0; r < 8; ++r) {
                product[(size_t)(b0 + mt * 16 + r + 8 * half) * DOUT + n0 + l16] =
                    acc[mt][r];
            }
    }
}

// ---------------------------------------------------------------------------
extern "C" void kernel_launch(void* const* d_in, const int* in_sizes, int n_in,
                              void* d_out, int out_size, void* d_ws, size_t ws_size,
                              hipStream_t stream) {
    const float* inputs = (const float*)d_in[0];  // (B, C, DS)
    const float* cent   = (const float*)d_in[1];  // (C, K, DS)
    const float* W      = (const float*)d_in[2];  // (DIN, DOUT)

    float* product = (float*)d_out;                               // B*DOUT
    float* nmse    = product + (size_t)B_TOT * DOUT;              // B*C
    float* codes   = nmse + (size_t)B_TOT * C_N;                  // B*C

    __bf16* Wt = (__bf16*)d_ws;                                   // DOUT*DIN bf16
    __bf16* cb = Wt + (size_t)DIN * DOUT;                         // C*K*DS bf16

    const int prep_total = DIN * DOUT + C_N * K_N * DS;
    dpq_prep_kernel<<<(prep_total + 255) / 256, 256, 0, stream>>>(W, cent, Wt, cb);

    const size_t lds_bytes = (size_t)ROWS_PER_BLOCK * ACT_STRIDE * sizeof(__bf16);
    dpq_main_kernel<<<B_TOT / ROWS_PER_BLOCK, 256, lds_bytes, stream>>>(
        inputs, cb, Wt, product, nmse, codes);
}
